// AggregationMambaBlock_84722524881201
// MI455X (gfx1250) — compile-verified, matrix operand order checked by reading, not run on hardware
//
#include <hip/hip_runtime.h>
#include <math.h>

// ---------------------------------------------------------------------------
// AggregationMambaBlock for MI455X (gfx1250, wave32, WMMA).
// 4 directional mamba blocks; GEMMs on v_wmma_f32_16x16x32_bf16, scan fused.
// ---------------------------------------------------------------------------

typedef __attribute__((ext_vector_type(16))) __bf16 v16bf;
typedef __attribute__((ext_vector_type(8)))  float  v8f;

#define NBLK   4
#define BATCH  4
#define LSEQ   1024
#define NTOK   (BATCH * LSEQ)   // 4096 tokens per block
#define DM     64               // d_model
#define DI     256              // d_inner
#define DST    16               // d_state
#define DTR    4                // dt_rank
#define XD     48               // padded xproj rows (36 -> 48)

__device__ __forceinline__ float silu_f(float x)     { return x / (1.f + __expf(-x)); }
__device__ __forceinline__ float softplus_f(float x) { return (x > 20.f) ? x : __logf(1.f + __expf(x)); }

// Load a 16x32 bf16 fragment from a row-major (rows x K) tile.
// Used for A (M x K) and for B supplied as N-major activations (N x K),
// which matches the ISA's "row striped across lanes" B layout (lane = column).
__device__ __forceinline__ v16bf load_frag_rm(const __bf16* p, int ld, int lane) {
    const int r    = lane & 15;
    const int half = lane >> 4;
    v16bf v;
#pragma unroll
    for (int i = 0; i < 16; ++i) {
        // lanes 0-15: K 0-7 (VGPR0-3) and 16-23 (VGPR4-7); lanes 16-31: K 8-15, 24-31
        const int kk = ((i >> 3) << 4) + (half << 3) + (i & 7);
        v[i] = p[r * ld + kk];
    }
    return v;
}

// ------------------------- packing / conversion ----------------------------

// Gather x into per-block scan order and convert to bf16: xseq[blk][tok][64]
__global__ void k_gather_x(const float* __restrict__ x, __bf16* __restrict__ xseq) {
    int i = blockIdx.x * blockDim.x + threadIdx.x;
    if (i >= NBLK * NTOK * DM) return;
    const int c     = i & (DM - 1);
    const int tok   = (i >> 6) & (NTOK - 1);
    const int b     = i >> 18;
    const int batch = tok >> 10;
    const int t     = tok & (LSEQ - 1);
    const int p = (b == 0) ? (LSEQ - 1 - t)
                : (b == 1) ? (LSEQ + t)
                : (b == 2) ? t
                           : (2 * LSEQ - 1 - t);
    xseq[i] = (__bf16)x[(batch * (2 * LSEQ) + p) * DM + c];
}

__global__ void k_cvt_inw(const float* __restrict__ w, __bf16* __restrict__ o) {
    int i = blockIdx.x * blockDim.x + threadIdx.x;
    if (i < NBLK * 2 * DI * DM) o[i] = (__bf16)w[i];
}

__global__ void k_cvt_xproj(const float* __restrict__ w, __bf16* __restrict__ o) {
    int i = blockIdx.x * blockDim.x + threadIdx.x;
    if (i >= NBLK * XD * DI) return;
    const int c = i & (DI - 1);
    const int r = (i >> 8) % XD;
    const int b = i / (XD * DI);
    const float v = (r < (DTR + 2 * DST)) ? w[(b * (DTR + 2 * DST) + r) * DI + c] : 0.f;
    o[i] = (__bf16)v;
}

__global__ void k_cvt_outw(const float* __restrict__ w, __bf16* __restrict__ o) {
    int i = blockIdx.x * blockDim.x + threadIdx.x;
    if (i < NBLK * DM * DI) o[i] = (__bf16)w[i];
}

// ------------------------- in-projection (WMMA) ----------------------------
// xz[512] = in_w(512x64) @ x_token(64).  One wave per 16x16 tile, K=64.
// Destination (xin_pre vs zbuf) is uniform per mt tile -> scalar select, no
// divergent stores.
__global__ void k_inproj(const __bf16* __restrict__ xseq, const __bf16* __restrict__ w,
                         float* __restrict__ xin_pre, float* __restrict__ zbuf) {
    const int nt = blockIdx.x;          // 256 token tiles
    const int mt = blockIdx.y;          // 32 channel tiles (512/16)
    const int b  = blockIdx.z;
    const int lane = threadIdx.x;
    const __bf16* A = w    + (size_t)b * 2 * DI * DM + mt * 16 * DM;       // 16 x 64
    const __bf16* B = xseq + ((size_t)b * NTOK + nt * 16) * DM;            // 16tok x 64
    v8f acc = {};
#pragma unroll
    for (int k0 = 0; k0 < DM; k0 += 32) {
        v16bf av = load_frag_rm(A + k0, DM, lane);
        v16bf bv = load_frag_rm(B + k0, DM, lane);
        acc = __builtin_amdgcn_wmma_f32_16x16x32_bf16(false, av, false, bv,
                                                      (short)0, acc, false, false);
    }
    // Uniform (workgroup-scalar) destination select: mt<16 -> xin_pre, else z.
    float* __restrict__ dst = (mt < 16) ? xin_pre : zbuf;
    const int chbase = ((mt < 16) ? mt : (mt - 16)) * 16;
    const int n = lane & 15, half = lane >> 4;
    const int tok = nt * 16 + n;
    float* row = dst + ((size_t)b * NTOK + tok) * DI + chbase + 8 * half;
#pragma unroll
    for (int j = 0; j < 8; ++j) row[j] = acc[j];
}

// --------------------- depthwise causal conv + SiLU ------------------------
__global__ void k_conv(const float* __restrict__ xin_pre,
                       const float* __restrict__ conv_w, const float* __restrict__ conv_b,
                       float* __restrict__ xin, __bf16* __restrict__ xin_bf) {
    int i = blockIdx.x * blockDim.x + threadIdx.x;
    if (i >= NBLK * NTOK * DI) return;
    const int ch  = i & (DI - 1);
    const int tok = (i >> 8) & (NTOK - 1);
    const int b   = i >> 20;
    const int t   = tok & (LSEQ - 1);
    float acc = conv_b[b * DI + ch];
#pragma unroll
    for (int j = 0; j < 4; ++j) {
        const int tt = t - 3 + j;
        if (tt >= 0)
            acc += conv_w[(b * DI + ch) * 4 + j] *
                   xin_pre[((size_t)b * NTOK + (tok - 3 + j)) * DI + ch];
    }
    const float s = silu_f(acc);
    xin[i]    = s;
    xin_bf[i] = (__bf16)s;
}

// ------------------------- x-projection (WMMA) -----------------------------
// xdbl[36] = xproj_w(36x256) @ xin(256); padded to 48 rows. K=256 -> 8 WMMAs.
__global__ void k_xproj(const __bf16* __restrict__ xin_bf, const __bf16* __restrict__ w,
                        float* __restrict__ xdbl) {
    const int nt = blockIdx.x;          // 256 token tiles
    const int mt = blockIdx.y;          // 3 row tiles (48/16)
    const int b  = blockIdx.z;
    const int lane = threadIdx.x;
    const __bf16* A = w      + (size_t)b * XD * DI + mt * 16 * DI;
    const __bf16* B = xin_bf + ((size_t)b * NTOK + nt * 16) * DI;
    v8f acc = {};
#pragma unroll
    for (int k0 = 0; k0 < DI; k0 += 32) {
        v16bf av = load_frag_rm(A + k0, DI, lane);
        v16bf bv = load_frag_rm(B + k0, DI, lane);
        acc = __builtin_amdgcn_wmma_f32_16x16x32_bf16(false, av, false, bv,
                                                      (short)0, acc, false, false);
    }
    const int n = lane & 15, half = lane >> 4;
    const int tok = nt * 16 + n;
#pragma unroll
    for (int j = 0; j < 8; ++j) {
        const int r = mt * 16 + j + 8 * half;
        xdbl[((size_t)b * NTOK + tok) * XD + r] = acc[j];
    }
}

// ---------------------- delta = softplus(dt @ dt_w.T + b) ------------------
__global__ void k_delta(const float* __restrict__ xdbl, const float* __restrict__ dt_w,
                        const float* __restrict__ dt_b, float* __restrict__ delta) {
    int i = blockIdx.x * blockDim.x + threadIdx.x;
    if (i >= NBLK * NTOK * DI) return;
    const int ch  = i & (DI - 1);
    const int tok = (i >> 8) & (NTOK - 1);
    const int b   = i >> 20;
    const float* dt = xdbl + ((size_t)b * NTOK + tok) * XD;
    float a = dt_b[b * DI + ch];
#pragma unroll
    for (int r = 0; r < DTR; ++r) a += dt[r] * dt_w[(b * DI + ch) * DTR + r];
    delta[i] = softplus_f(a);
}

// --------------------------- selective scan --------------------------------
// One workgroup per (blk,batch); one lane per channel; 16 states in registers.
// dA/dBx computed on the fly; Bs/Cs double-buffered in LDS so only one
// barrier per timestep is needed; output gated with silu(z), stored bf16.
__global__ void __launch_bounds__(DI)
k_scan(const float* __restrict__ xdbl, const float* __restrict__ delta,
       const float* __restrict__ xin, const float* __restrict__ A_log,
       const float* __restrict__ D_param, const float* __restrict__ zbuf,
       __bf16* __restrict__ ygated) {
    const int w = blockIdx.x;           // 16 = 4 blk x 4 batch
    const int b = w >> 2, batch = w & 3;
    const int d = threadIdx.x;          // channel 0..255
    __shared__ float BC[2][2 * DST];    // double-buffered Bs[16] | Cs[16]
    float A[DST], h[DST];
#pragma unroll
    for (int s = 0; s < DST; ++s) {
        A[s] = -__expf(A_log[((size_t)b * DI + d) * DST + s]);
        h[s] = 0.f;
    }
    const float Dp = D_param[b * DI + d];
    for (int t = 0; t < LSEQ; ++t) {
        const size_t gi  = (size_t)b * NTOK + batch * LSEQ + t;
        const int    buf = t & 1;
        if (d < 2 * DST) BC[buf][d] = xdbl[gi * XD + DTR + d];
        __syncthreads();                 // orders buf writes before buf reads;
                                         // next-iter writes go to the other buf
        const float de = delta[gi * DI + d];
        const float xv = xin[gi * DI + d];
        float y = xv * Dp;
#pragma unroll
        for (int s = 0; s < DST; ++s) {
            const float dA = __expf(de * A[s]);
            h[s] = dA * h[s] + de * BC[buf][s] * xv;
            y += h[s] * BC[buf][DST + s];
        }
        const float zz = zbuf[gi * DI + d];
        ygated[gi * DI + d] = (__bf16)(y * silu_f(zz));
    }
}

// ------------- out-projection (WMMA) + residual + tanh epilogue ------------
__global__ void k_outproj(const __bf16* __restrict__ yg, const __bf16* __restrict__ ow,
                          const float* __restrict__ x,
                          const float* __restrict__ alpha, const float* __restrict__ beta,
                          const float* __restrict__ gamma, const float* __restrict__ beta1,
                          float* __restrict__ out) {
    const int nt = blockIdx.x;          // 256 token tiles
    const int mt = blockIdx.y;          // 4 channel tiles (64/16)
    const int b  = blockIdx.z;
    const int lane = threadIdx.x;
    const __bf16* A = ow + (size_t)b * DM * DI + mt * 16 * DI;
    const __bf16* B = yg + ((size_t)b * NTOK + nt * 16) * DI;
    v8f acc = {};
#pragma unroll
    for (int k0 = 0; k0 < DI; k0 += 32) {
        v16bf av = load_frag_rm(A + k0, DI, lane);
        v16bf bv = load_frag_rm(B + k0, DI, lane);
        acc = __builtin_amdgcn_wmma_f32_16x16x32_bf16(false, av, false, bv,
                                                      (short)0, acc, false, false);
    }
    const float al = alpha[0], ga = gamma[0];
    const int n = lane & 15, half = lane >> 4;
    const int tok   = nt * 16 + n;
    const int batch = tok >> 10;
    const int t     = tok & (LSEQ - 1);
    const int p = (b == 0) ? (LSEQ - 1 - t)
                : (b == 1) ? (LSEQ + t)
                : (b == 2) ? t
                           : (2 * LSEQ - 1 - t);
    const int coff = (b >= 2) ? DM : 0;
#pragma unroll
    for (int j = 0; j < 8; ++j) {
        const int m = mt * 16 + j + 8 * half;   // out channel 0..63
        const float v = acc[j] + x[((size_t)batch * (2 * LSEQ) + p) * DM + m];
        const int c = coff + m;
        out[((size_t)batch * (2 * LSEQ) + p) * (2 * DM) + c] =
            ga * tanhf(al * v + beta1[c]) + beta[c];
    }
}

// ---------------------------------------------------------------------------

extern "C" void kernel_launch(void* const* d_in, const int* in_sizes, int n_in,
                              void* d_out, int out_size, void* d_ws, size_t ws_size,
                              hipStream_t stream) {
    (void)in_sizes; (void)n_in; (void)out_size; (void)ws_size;
    const float* x       = (const float*)d_in[0];
    const float* in_w    = (const float*)d_in[1];
    const float* conv_w  = (const float*)d_in[2];
    const float* conv_b  = (const float*)d_in[3];
    const float* xproj_w = (const float*)d_in[4];
    const float* dt_w    = (const float*)d_in[5];
    const float* dt_b    = (const float*)d_in[6];
    const float* A_log   = (const float*)d_in[7];
    const float* D_param = (const float*)d_in[8];
    const float* out_w   = (const float*)d_in[9];
    const float* dy_alpha = (const float*)d_in[10];
    const float* dy_beta  = (const float*)d_in[11];
    const float* dy_gamma = (const float*)d_in[12];
    const float* dy_beta1 = (const float*)d_in[13];
    float* out = (float*)d_out;

    char* ws = (char*)d_ws;
    size_t off = 0;
    auto take = [&](size_t bytes) -> void* {
        void* p = ws + off;
        off += (bytes + 255) & ~(size_t)255;
        return p;
    };
    __bf16* xseq_bf   = (__bf16*)take((size_t)NBLK * NTOK * DM * 2);
    __bf16* inw_bf    = (__bf16*)take((size_t)NBLK * 2 * DI * DM * 2);
    __bf16* xprojp_bf = (__bf16*)take((size_t)NBLK * XD * DI * 2);
    __bf16* outw_bf   = (__bf16*)take((size_t)NBLK * DM * DI * 2);
    float*  xin_pre   = (float*) take((size_t)NBLK * NTOK * DI * 4);
    float*  zbuf      = (float*) take((size_t)NBLK * NTOK * DI * 4);
    float*  xin       = (float*) take((size_t)NBLK * NTOK * DI * 4);
    __bf16* xin_bf    = (__bf16*)take((size_t)NBLK * NTOK * DI * 2);
    float*  xdbl      = (float*) take((size_t)NBLK * NTOK * XD * 4);
    float*  delta     = (float*) take((size_t)NBLK * NTOK * DI * 4);
    __bf16* ygated    = (__bf16*)take((size_t)NBLK * NTOK * DI * 2);

    // Pack / convert
    {
        int n = NBLK * NTOK * DM;
        k_gather_x<<<(n + 255) / 256, 256, 0, stream>>>(x, xseq_bf);
    }
    {
        int n = NBLK * 2 * DI * DM;
        k_cvt_inw<<<(n + 255) / 256, 256, 0, stream>>>(in_w, inw_bf);
    }
    {
        int n = NBLK * XD * DI;
        k_cvt_xproj<<<(n + 255) / 256, 256, 0, stream>>>(xproj_w, xprojp_bf);
    }
    {
        int n = NBLK * DM * DI;
        k_cvt_outw<<<(n + 255) / 256, 256, 0, stream>>>(out_w, outw_bf);
    }

    // Token-parallel GEMMs + pointwise
    k_inproj<<<dim3(NTOK / 16, 2 * DI / 16, NBLK), 32, 0, stream>>>(xseq_bf, inw_bf, xin_pre, zbuf);
    {
        int n = NBLK * NTOK * DI;
        k_conv<<<(n + 255) / 256, 256, 0, stream>>>(xin_pre, conv_w, conv_b, xin, xin_bf);
    }
    k_xproj<<<dim3(NTOK / 16, XD / 16, NBLK), 32, 0, stream>>>(xin_bf, xprojp_bf, xdbl);
    {
        int n = NBLK * NTOK * DI;
        k_delta<<<(n + 255) / 256, 256, 0, stream>>>(xdbl, dt_w, dt_b, delta);
    }

    // Sequential selective scan (latency-bound core)
    k_scan<<<NBLK * BATCH, DI, 0, stream>>>(xdbl, delta, xin, A_log, D_param, zbuf, ygated);

    // Out-projection + residual + tanh epilogue
    k_outproj<<<dim3(NTOK / 16, DM / 16, NBLK), 32, 0, stream>>>(
        ygated, outw_bf, x, dy_alpha, dy_beta, dy_gamma, dy_beta1, out);
}
